// BlockchainGNN_223338299944
// MI455X (gfx1250) — matrix operand order repaired and checked by reading, not compiled, not run on hardware
//
#include <hip/hip_runtime.h>
#include <hip/hip_bf16.h>
#include <stdint.h>

typedef __attribute__((ext_vector_type(16))) _Float16 v16h;
typedef __attribute__((ext_vector_type(8)))  float    v8f;

#define HEADS    4
#define OUTC     32
#define BN_EPS   1e-5f
#define NEGSLOPE 0.2f
#define NT       4      // n-tiles (16 cols each) per wave -> 16x64 output tile

// ---------- order-preserving float<->uint encoding for atomicMax ----------
__device__ __forceinline__ unsigned int enc_f32(float f) {
    unsigned int u = __float_as_uint(f);
    return (u & 0x80000000u) ? ~u : (u | 0x80000000u);
}
__device__ __forceinline__ float dec_f32(unsigned int u) {
    return (u & 0x80000000u) ? __uint_as_float(u ^ 0x80000000u)
                             : __uint_as_float(~u);
}
__device__ __forceinline__ float lrelu(float v) {
    return v > 0.0f ? v : NEGSLOPE * v;
}

// ---------- pack fp32 weights [K x ncol] into wave32 B-fragment layout ----------
// Layout: Bp[((tile*kSteps + ks)*32 + lane)*16 + e] = (f16)W[k][c]
//   half = lane>>4, lr = lane&15, k = ks*32 + half*16 + e, c = tile*16 + lr
__global__ void pack_b_kernel(const float* __restrict__ W, _Float16* __restrict__ Bp,
                              int K, int ncol) {
    int t = blockIdx.x * blockDim.x + threadIdx.x;
    int kSteps = K >> 5;
    int total = (ncol >> 4) * kSteps * 512;
    if (t >= total) return;
    int e    = t & 15;
    int lane = (t >> 4) & 31;
    int rest = t >> 9;
    int ks   = rest % kSteps;
    int tile = rest / kSteps;
    int half = lane >> 4, lr = lane & 15;
    int k = (ks << 5) + half * 16 + e;
    int c = (tile << 4) + lr;
    Bp[t] = (_Float16)W[(size_t)k * ncol + c];
}

// ---------- in-degree count ----------
__global__ void count_deg(const int* __restrict__ dst, int E, float* __restrict__ cnt) {
    int t = blockIdx.x * blockDim.x + threadIdx.x;
    if (t < E) atomicAdd(&cnt[dst[t]], 1.0f);
}

// ---------- edge scatter-add: agg[dst] += x[src] ----------
__global__ void scatter_add(const float* __restrict__ x, const int* __restrict__ src,
                            const int* __restrict__ dst, int E, int shift,
                            float* __restrict__ agg) {
    int t = blockIdx.x * blockDim.x + threadIdx.x;
    int total = E << shift;
    if (t >= total) return;
    int e = t >> shift;
    int c = t & ((1 << shift) - 1);
    int C = 1 << shift;
    atomicAdd(&agg[(size_t)dst[e] * C + c], x[(size_t)src[e] * C + c]);
}

// ---------- mean = agg / max(cnt,1) ----------
__global__ void mean_div(float* __restrict__ agg, const float* __restrict__ cnt,
                         int n, int shift) {
    int t = blockIdx.x * blockDim.x + threadIdx.x;
    if (t >= (n << shift)) return;
    int node = t >> shift;
    agg[t] /= fmaxf(cnt[node], 1.0f);
}

// ---------- A fragment load: fp32 row -> f16 fragment (contiguous k, matches packed B) ----------
__device__ __forceinline__ v16h load_a_frag(const float* __restrict__ A, size_t rowOff,
                                            int ka) {
    const float4* p0 = (const float4*)(A + rowOff + ka);
    const float4* p1 = (const float4*)(A + rowOff + ka + 8);
    float4 f0 = p0[0], f1 = p0[1];
    float4 f2 = p1[0], f3 = p1[1];
    v16h a;
    a[0]  = (_Float16)f0.x; a[1]  = (_Float16)f0.y; a[2]  = (_Float16)f0.z; a[3]  = (_Float16)f0.w;
    a[4]  = (_Float16)f1.x; a[5]  = (_Float16)f1.y; a[6]  = (_Float16)f1.z; a[7]  = (_Float16)f1.w;
    a[8]  = (_Float16)f2.x; a[9]  = (_Float16)f2.y; a[10] = (_Float16)f2.z; a[11] = (_Float16)f2.w;
    a[12] = (_Float16)f3.x; a[13] = (_Float16)f3.y; a[14] = (_Float16)f3.z; a[15] = (_Float16)f3.w;
    return a;
}

// ---------- WMMA GEMM: out = f( A1@W1 [+ A2@W2] + bias ; BN ; +resid ; relu ) ----------
// A: fp32 [nrows x K] (lda), Bp: packed f16 fragments, out: fp32 [nrows x ncol]
// Each wave computes a 16 x (16*NT) output tile. ncol must be a multiple of 64.
__global__ __launch_bounds__(256)
void wmma_gemm_kernel(const float* __restrict__ A1, const _Float16* __restrict__ B1p,
                      const float* __restrict__ A2, const _Float16* __restrict__ B2p,
                      const float* __restrict__ bias,
                      const float* __restrict__ bn_g, const float* __restrict__ bn_b,
                      const float* __restrict__ bn_m, const float* __restrict__ bn_v,
                      const float* __restrict__ resid,
                      float* __restrict__ out,
                      int nrows, int K, int ncol, int lda, int do_relu) {
    const int groupsN = ncol >> 6;                // 64-wide column groups
    const int tilesM  = (nrows + 15) >> 4;
    const int wave = blockIdx.x * (blockDim.x >> 5) + (threadIdx.x >> 5);
    if (wave >= tilesM * groupsN) return;
    const int tm = wave / groupsN;
    const int tg = wave % groupsN;
    const int lane = threadIdx.x & 31;
    const int half = lane >> 4;
    const int lr   = lane & 15;
    const int m0 = tm << 4;
    int row = m0 + lr;
    if (row >= nrows) row = nrows - 1;            // clamped rows feed only unstored D rows
    const size_t rowOff = (size_t)row * lda;
    const int kSteps = K >> 5;

    v8f acc[NT];
    #pragma unroll
    for (int tt = 0; tt < NT; ++tt) acc[tt] = (v8f){};

    const v16h* B1v = (const v16h*)B1p;
    const v16h* B2v = (const v16h*)B2p;

    for (int ks = 0; ks < kSteps; ++ks) {
        const int ka = (ks << 5) + half * 16;

        // issue ALL independent loads first so the compiler can clause them
        // and count loadcnt down across the WMMA chain (pipelined latency)
        v16h a1 = load_a_frag(A1, rowOff, ka);
        v16h b1f[NT];
        #pragma unroll
        for (int tt = 0; tt < NT; ++tt)
            b1f[tt] = B1v[((size_t)(tg * NT + tt) * kSteps + ks) * 32 + lane];
        #pragma unroll
        for (int tt = 0; tt < NT; ++tt)
            acc[tt] = __builtin_amdgcn_wmma_f32_16x16x32_f16(false, a1, false, b1f[tt],
                                                             (short)0, acc[tt], false, false);
        if (A2) {
            v16h a2 = load_a_frag(A2, rowOff, ka);
            v16h b2f[NT];
            #pragma unroll
            for (int tt = 0; tt < NT; ++tt)
                b2f[tt] = B2v[((size_t)(tg * NT + tt) * kSteps + ks) * 32 + lane];
            #pragma unroll
            for (int tt = 0; tt < NT; ++tt)
                acc[tt] = __builtin_amdgcn_wmma_f32_16x16x32_f16(false, a2, false, b2f[tt],
                                                                 (short)0, acc[tt], false, false);
        }
    }

    // epilogue: per-lane channel is fixed per tt; node varies with acc index
    #pragma unroll
    for (int tt = 0; tt < NT; ++tt) {
        const int col = ((tg * NT + tt) << 4) + lr;
        float addb = bias ? bias[col] : 0.0f;
        float g = 1.f, bb = 0.f, mm = 0.f, inv = 1.f;
        if (bn_g) {
            g = bn_g[col]; bb = bn_b[col]; mm = bn_m[col];
            inv = rsqrtf(bn_v[col] + BN_EPS);
        }
        #pragma unroll
        for (int r = 0; r < 8; ++r) {
            int node = m0 + r + half * 8;         // D layout: lanes 16-31 hold M=8..15
            if (node < nrows) {
                float v = acc[tt][r] + addb;
                if (bn_g)  v = (v - mm) * inv * g + bb;
                if (resid) v += resid[(size_t)node * ncol + col];
                if (do_relu) v = fmaxf(v, 0.0f);
                out[(size_t)node * ncol + col] = v;
            }
        }
    }
}

// ---------- GAT: per-node attention logits ----------
__global__ void gat_att_kernel(const float* __restrict__ hg,
                               const float* __restrict__ att_src,
                               const float* __restrict__ att_dst,
                               float* __restrict__ asrc, float* __restrict__ adst, int n) {
    int t = blockIdx.x * blockDim.x + threadIdx.x;
    if (t >= n * HEADS) return;
    int node = t >> 2, h = t & 3;
    const float* hp = hg + (size_t)node * (HEADS * OUTC) + h * OUTC;
    float s = 0.f, d = 0.f;
    #pragma unroll
    for (int c = 0; c < OUTC; ++c) {
        s += hp[c] * att_src[h * OUTC + c];
        d += hp[c] * att_dst[h * OUTC + c];
    }
    asrc[t] = s;
    adst[t] = d;
}

// ---------- GAT: segment max (edges + self loops) ----------
__global__ void gat_max_kernel(const int* __restrict__ src, const int* __restrict__ dst,
                               int E, int n, const float* __restrict__ asrc,
                               const float* __restrict__ adst, unsigned int* __restrict__ menc) {
    int t = blockIdx.x * blockDim.x + threadIdx.x;
    if (t >= (E + n) * HEADS) return;
    int idx = t >> 2, h = t & 3;
    int s, d;
    if (idx < E) { s = src[idx]; d = dst[idx]; } else { s = d = idx - E; }
    float e = lrelu(asrc[s * HEADS + h] + adst[d * HEADS + h]);
    atomicMax(&menc[d * HEADS + h], enc_f32(e));
}

// ---------- GAT: softmax denominator ----------
__global__ void gat_den_kernel(const int* __restrict__ src, const int* __restrict__ dst,
                               int E, int n, const float* __restrict__ asrc,
                               const float* __restrict__ adst,
                               const unsigned int* __restrict__ menc, float* __restrict__ den) {
    int t = blockIdx.x * blockDim.x + threadIdx.x;
    if (t >= (E + n) * HEADS) return;
    int idx = t >> 2, h = t & 3;
    int s, d;
    if (idx < E) { s = src[idx]; d = dst[idx]; } else { s = d = idx - E; }
    float e = lrelu(asrc[s * HEADS + h] + adst[d * HEADS + h]);
    float m = dec_f32(menc[d * HEADS + h]);
    atomicAdd(&den[d * HEADS + h], expf(e - m));
}

// ---------- GAT: weighted aggregation (unnormalized) ----------
__global__ void gat_agg_kernel(const int* __restrict__ src, const int* __restrict__ dst,
                               int E, int n, const float* __restrict__ asrc,
                               const float* __restrict__ adst,
                               const unsigned int* __restrict__ menc,
                               const float* __restrict__ hg, float* __restrict__ acc) {
    int t = blockIdx.x * blockDim.x + threadIdx.x;
    int total = (E + n) * (HEADS * OUTC);
    if (t >= total) return;
    int idx = t >> 7;          // 128 = HEADS*OUTC
    int hc  = t & 127;
    int h   = hc >> 5;
    int s, d;
    if (idx < E) { s = src[idx]; d = dst[idx]; } else { s = d = idx - E; }
    float e = lrelu(asrc[s * HEADS + h] + adst[d * HEADS + h]);
    float m = dec_f32(menc[d * HEADS + h]);
    float ex = expf(e - m);
    atomicAdd(&acc[(size_t)d * 128 + hc], ex * hg[(size_t)s * 128 + hc]);
}

// ---------- GAT: normalize, head-mean, bias -> embeddings ----------
__global__ void gat_final_kernel(const float* __restrict__ acc, const float* __restrict__ den,
                                 const float* __restrict__ bias, float* __restrict__ emb, int n) {
    int t = blockIdx.x * blockDim.x + threadIdx.x;
    if (t >= n * OUTC) return;
    int node = t >> 5, c = t & 31;
    float s = 0.f;
    #pragma unroll
    for (int h = 0; h < HEADS; ++h)
        s += acc[(size_t)node * 128 + h * OUTC + c] / den[node * HEADS + h];
    emb[(size_t)node * OUTC + c] = 0.25f * s + bias[c];
}

// ---------- MLP second layer: logits ----------
__global__ void logits_kernel(const float* __restrict__ hid, const float* __restrict__ w2,
                              const float* __restrict__ b2, float* __restrict__ out, int n) {
    int t = blockIdx.x * blockDim.x + threadIdx.x;
    if (t >= n) return;
    float s = b2[0];
    #pragma unroll 8
    for (int j = 0; j < 64; ++j) s += hid[(size_t)t * 64 + j] * w2[j];
    out[t] = s;
}

static inline unsigned int cdiv(long long a, long long b) { return (unsigned int)((a + b - 1) / b); }

extern "C" void kernel_launch(void* const* d_in, const int* in_sizes, int n_in,
                              void* d_out, int out_size, void* d_ws, size_t ws_size,
                              hipStream_t stream) {
    const float* x        = (const float*)d_in[0];
    const int*   ei       = (const int*)d_in[1];
    const float* s1_wl    = (const float*)d_in[2];
    const float* s1_bl    = (const float*)d_in[3];
    const float* s1_wr    = (const float*)d_in[4];
    const float* bn1_g    = (const float*)d_in[5];
    const float* bn1_b    = (const float*)d_in[6];
    const float* bn1_m    = (const float*)d_in[7];
    const float* bn1_v    = (const float*)d_in[8];
    const float* s2_wl    = (const float*)d_in[9];
    const float* s2_bl    = (const float*)d_in[10];
    const float* s2_wr    = (const float*)d_in[11];
    const float* bn2_g    = (const float*)d_in[12];
    const float* bn2_b    = (const float*)d_in[13];
    const float* bn2_m    = (const float*)d_in[14];
    const float* bn2_v    = (const float*)d_in[15];
    const float* gat_w    = (const float*)d_in[16];
    const float* att_src  = (const float*)d_in[17];
    const float* att_dst  = (const float*)d_in[18];
    const float* gat_bias = (const float*)d_in[19];
    const float* cls_w1   = (const float*)d_in[20];
    const float* cls_b1   = (const float*)d_in[21];
    const float* cls_w2   = (const float*)d_in[22];
    const float* cls_b2   = (const float*)d_in[23];

    const int N = in_sizes[0] / 64;     // 50000
    const int E = in_sizes[1] / 2;      // 800000
    const int* src = ei;
    const int* dst = ei + E;

    // ---- workspace layout (~80 MB -> resident in 192 MB L2) ----
    float* AGG  = (float*)d_ws;                    // [N,128] scatter accumulator / GAT acc
    float* H    = AGG + (size_t)N * 128;           // [N,128] layer1 out / GAT hg
    float* H2   = H   + (size_t)N * 128;           // [N,128] layer2 out / MLP hidden
    float* cnt  = H2  + (size_t)N * 128;           // [N]
    float* asrc = cnt + (size_t)N;                 // [N,4]
    float* adst = asrc + (size_t)N * 4;            // [N,4]
    unsigned int* menc = (unsigned int*)(adst + (size_t)N * 4);  // [N,4]
    float* den  = (float*)(menc + (size_t)N * 4);  // [N,4]
    uintptr_t wbase = (uintptr_t)(den + (size_t)N * 4);
    wbase = (wbase + 31) & ~(uintptr_t)31;         // 32B-align packed B fragments
    _Float16* w_s1l = (_Float16*)wbase;
    _Float16* w_s1r = w_s1l + 64 * 128;
    _Float16* w_s2l = w_s1r + 64 * 128;
    _Float16* w_s2r = w_s2l + 128 * 128;
    _Float16* w_gat = w_s2r + 128 * 128;
    _Float16* w_c1  = w_gat + 128 * 128;

    float* emb    = (float*)d_out;              // [N,32]
    float* logits = emb + (size_t)N * OUTC;     // [N]

    const int B = 256;

    // ---- pack weights into f16 B-fragment layout (once per launch) ----
    pack_b_kernel<<<cdiv(64 * 128, B), B, 0, stream>>>(s1_wl, w_s1l, 64, 128);
    pack_b_kernel<<<cdiv(64 * 128, B), B, 0, stream>>>(s1_wr, w_s1r, 64, 128);
    pack_b_kernel<<<cdiv(128 * 128, B), B, 0, stream>>>(s2_wl, w_s2l, 128, 128);
    pack_b_kernel<<<cdiv(128 * 128, B), B, 0, stream>>>(s2_wr, w_s2r, 128, 128);
    pack_b_kernel<<<cdiv(128 * 128, B), B, 0, stream>>>(gat_w, w_gat, 128, 128);
    pack_b_kernel<<<cdiv(32 * 64, B), B, 0, stream>>>(cls_w1, w_c1, 32, 64);

    // ---- in-degree (shared by both SAGE layers) ----
    hipMemsetAsync(cnt, 0, (size_t)N * sizeof(float), stream);
    count_deg<<<cdiv(E, B), B, 0, stream>>>(dst, E, cnt);

    // ================= SAGE layer 1 (64 -> 128) =================
    hipMemsetAsync(AGG, 0, (size_t)N * 64 * sizeof(float), stream);
    scatter_add<<<cdiv((long long)E * 64, B), B, 0, stream>>>(x, src, dst, E, 6, AGG);
    mean_div<<<cdiv((long long)N * 64, B), B, 0, stream>>>(AGG, cnt, N, 6);
    {
        int waves = ((N + 15) / 16) * (128 / 64);
        wmma_gemm_kernel<<<cdiv(waves, 8), B, 0, stream>>>(
            AGG, w_s1l, x, w_s1r, s1_bl,
            bn1_g, bn1_b, bn1_m, bn1_v, nullptr, H, N, 64, 128, 64, 1);
    }

    // ================= SAGE layer 2 (128 -> 128, residual) =================
    hipMemsetAsync(AGG, 0, (size_t)N * 128 * sizeof(float), stream);
    scatter_add<<<cdiv((long long)E * 128, B), B, 0, stream>>>(H, src, dst, E, 7, AGG);
    mean_div<<<cdiv((long long)N * 128, B), B, 0, stream>>>(AGG, cnt, N, 7);
    {
        int waves = ((N + 15) / 16) * (128 / 64);
        wmma_gemm_kernel<<<cdiv(waves, 8), B, 0, stream>>>(
            AGG, w_s2l, H, w_s2r, s2_bl,
            bn2_g, bn2_b, bn2_m, bn2_v, H, H2, N, 128, 128, 128, 1);
    }

    // ================= GAT (128 -> 4 heads x 32) =================
    float* HG = H;    // H free after residual use -> reuse for h2 @ gat_w
    {
        int waves = ((N + 15) / 16) * (128 / 64);
        wmma_gemm_kernel<<<cdiv(waves, 8), B, 0, stream>>>(
            H2, w_gat, nullptr, nullptr, nullptr,
            nullptr, nullptr, nullptr, nullptr, nullptr, HG, N, 128, 128, 128, 0);
    }
    gat_att_kernel<<<cdiv((long long)N * HEADS, B), B, 0, stream>>>(HG, att_src, att_dst, asrc, adst, N);
    hipMemsetAsync(menc, 0, (size_t)N * HEADS * sizeof(unsigned int), stream);
    gat_max_kernel<<<cdiv((long long)(E + N) * HEADS, B), B, 0, stream>>>(src, dst, E, N, asrc, adst, menc);
    hipMemsetAsync(den, 0, (size_t)N * HEADS * sizeof(float), stream);
    gat_den_kernel<<<cdiv((long long)(E + N) * HEADS, B), B, 0, stream>>>(src, dst, E, N, asrc, adst, menc, den);
    float* GACC = AGG;   // AGG free after layer-2 GEMM -> reuse
    hipMemsetAsync(GACC, 0, (size_t)N * 128 * sizeof(float), stream);
    gat_agg_kernel<<<cdiv((long long)(E + N) * 128, B), B, 0, stream>>>(src, dst, E, N, asrc, adst, menc, HG, GACC);
    gat_final_kernel<<<cdiv((long long)N * OUTC, B), B, 0, stream>>>(GACC, den, gat_bias, emb, N);

    // ================= MLP head (32 -> 64 -> 1) =================
    float* MLPH = H2;  // H2 free after GAT GEMM -> reuse
    {
        int waves = ((N + 15) / 16) * (64 / 64);
        wmma_gemm_kernel<<<cdiv(waves, 8), B, 0, stream>>>(
            emb, w_c1, nullptr, nullptr, cls_b1,
            nullptr, nullptr, nullptr, nullptr, nullptr, MLPH, N, 32, 64, 32, 1);
    }
    logits_kernel<<<cdiv(N, B), B, 0, stream>>>(MLPH, cls_w2, cls_b2, logits, N);
}
